// CausalSelfAttention_89378269429885
// MI455X (gfx1250) — matrix operand order repaired and checked by reading, not compiled, not run on hardware
//
#include <hip/hip_runtime.h>
#include <hip/hip_bf16.h>

// ---------------------------------------------------------------------------
// Problem constants (fixed by the reference setup)
// ---------------------------------------------------------------------------
constexpr int BB  = 2;      // batch
constexpr int SS  = 2048;   // sequence
constexpr int DD  = 2048;   // model dim
constexpr int NH  = 16;     // query heads
constexpr int NKV = 4;      // kv heads
constexpr int HD  = 128;    // head dim
constexpr float EPS_RMS = 1.1920929e-07f;
constexpr float ATT_SCALE = 0.08838834764831845f; // 1/sqrt(128)

typedef __attribute__((ext_vector_type(16))) __bf16 v16bf;
typedef __attribute__((ext_vector_type(8)))  float  v8f;
typedef __attribute__((ext_vector_type(4)))  unsigned int v4u;
typedef __attribute__((ext_vector_type(8)))  int v8i_;
typedef __attribute__((ext_vector_type(4)))  int v4i_;

union BF16x16 { v16bf v; unsigned short u[16]; unsigned int w[8]; };

#if defined(__has_builtin)
#if __has_builtin(__builtin_amdgcn_tensor_load_to_lds) && \
    __has_builtin(__builtin_amdgcn_s_wait_tensorcnt)
#define HAVE_TDM 1
#endif
#endif
#ifndef HAVE_TDM
#define HAVE_TDM 0
#endif

__device__ inline unsigned short f2bf(float f) {
  unsigned int u = __float_as_uint(f);
  unsigned int r = u + 0x7FFFu + ((u >> 16) & 1u);   // round-to-nearest-even
  return (unsigned short)(r >> 16);
}

__device__ inline v8f wmma_bf16(v16bf a, v16bf b, v8f c) {
  // (neg_a, A, neg_b, B, c_mod, C, reuse_a, reuse_b)
  return __builtin_amdgcn_wmma_f32_16x16x32_bf16(false, a, false, b, (short)0, c,
                                                 false, false);
}

// A fragment (16x32 bf16): lane holds row = lane&15; VGPR v holds K pair
//   k = (v>=4?16:0) + (lane>=16?8:0) + 2*(v&3) + {0,1}
__device__ inline v16bf lds_a_frag(const unsigned short* base, int stride, int lane) {
  BF16x16 f;
  const unsigned short* p = base + (lane & 15) * stride;
  int koff = (lane >> 4) << 3;
#pragma unroll
  for (int v = 0; v < 8; ++v) {
    int k = ((v & 4) << 2) + koff + ((v & 3) << 1);
    f.w[v] = *(const unsigned int*)(p + k);
  }
  return f.v;
}

// B fragment (32x16 bf16): lane holds col n = lane&15; VGPR j holds K pair
//   k = (j>=4?16:0) + (lane>=16?8:0) + 2*(j&3) + {0,1}
__device__ inline v16bf lds_b_frag(const unsigned short* base, int stride, int lane) {
  BF16x16 f;
  const unsigned short* p = base + (lane & 15) * stride + ((lane >> 4) << 3);
#pragma unroll
  for (int j = 0; j < 8; ++j) f.w[j] = *(const unsigned int*)(p + ((j & 4) << 2) + ((j & 3) << 1));
  return f.v;
}

#if HAVE_TDM
// ---------------------------------------------------------------------------
// Issue one TDM 2D tile load (global -> LDS) per cdna5 D# layout:
//  group0: [1:0]=count=1, [63:32]=lds_addr, [120:64]=global_addr, [127:126]=2
//  group1: [17:16]=data_size(1=2B), [20]=pad_enable, [24:22]=pad_interval,
//          [31:25]=pad_amount, [79:48]=tensor_dim0, [111:80]=tensor_dim1,
//          [127:112]=tile_dim0, [143:128]=tile_dim1, [207:160]=dim0_stride
//  Units are data_size elements. pad_interval: 5 -> every 64 DW, 4 -> every 32 DW;
//  pad_amount: 1 -> 2 DWORDs.
// ---------------------------------------------------------------------------
__device__ inline void tdm_load_2d(unsigned lds_off, const void* gaddr,
                                   unsigned tensor_d0, unsigned tensor_d1,
                                   unsigned stride0, unsigned tile_d0,
                                   unsigned tile_d1, unsigned pad_interval,
                                   unsigned pad_amount) {
  unsigned long long ga = (unsigned long long)gaddr;
  v4u g0;
  g0[0] = 1u;                      // count=1, user descriptor
  g0[1] = lds_off;                 // LDS byte address
  g0[2] = (unsigned)ga;            // global addr [95:64]
  g0[3] = (unsigned)(ga >> 32) | (2u << 30);  // addr hi | type=2 (image)
  v8i_ g1;
  g1[0] = (int)((1u << 16) | (1u << 20) | (pad_interval << 22) | (pad_amount << 25));
  g1[1] = (int)(tensor_d0 << 16);                       // tensor_dim0 lo16
  g1[2] = (int)((tensor_d0 >> 16) | (tensor_d1 << 16)); // dim0 hi | dim1 lo
  g1[3] = (int)((tensor_d1 >> 16) | (tile_d0 << 16));   // dim1 hi | tile_dim0
  g1[4] = (int)tile_d1;                                 // tile_dim1 (tile_dim2=0)
  g1[5] = (int)stride0;                                 // tensor_dim0_stride lo32
  g1[6] = 0;
  g1[7] = 0;
  v4i_ z4 = {0, 0, 0, 0};
#if __clang_major__ >= 23
  v8i_ z8 = {0, 0, 0, 0, 0, 0, 0, 0};
  __builtin_amdgcn_tensor_load_to_lds(g0, g1, z4, z4, z8, 0);
#else
  __builtin_amdgcn_tensor_load_to_lds(g0, g1, z4, z4, 0);
#endif
}
#endif  // HAVE_TDM

// ---------------------------------------------------------------------------
// GEMM: C[M,N] (f32) = A[M,K] @ W[N,K]^T, bf16 WMMA, f32 accumulate.
// Tiles 128x128x32, 256 threads (8 waves), wave computes 32x64.
// ---------------------------------------------------------------------------
template <bool ABF16>
__global__ __launch_bounds__(256) void gemm_kernel(const void* __restrict__ Ap,
                                                   const float* __restrict__ Wt,
                                                   float* __restrict__ C,
                                                   int M, int N, int K) {
  __shared__ unsigned short As[128][34];
  __shared__ unsigned short Bs[128][34];
  const int m0 = blockIdx.y * 128, n0 = blockIdx.x * 128;
  const int tid = threadIdx.x, w = tid >> 5, lane = tid & 31;
  const int wm = w & 3, wn = w >> 2;
  const float* Af = (const float*)Ap;
  const unsigned short* Ah = (const unsigned short*)Ap;

  v8f acc[2][4];
#pragma unroll
  for (int mi = 0; mi < 2; ++mi)
#pragma unroll
    for (int ni = 0; ni < 4; ++ni)
#pragma unroll
      for (int r = 0; r < 8; ++r) acc[mi][ni][r] = 0.f;

  for (int k0 = 0; k0 < K; k0 += 32) {
    __syncthreads();
#pragma unroll
    for (int i = 0; i < 16; ++i) {
      int e = tid + i * 256, r = e >> 5, c = e & 31;
      unsigned short av;
      if (ABF16) av = Ah[(size_t)(m0 + r) * K + k0 + c];
      else       av = f2bf(Af[(size_t)(m0 + r) * K + k0 + c]);
      As[r][c] = av;
      Bs[r][c] = f2bf(Wt[(size_t)(n0 + r) * K + k0 + c]);
    }
    if (k0 + 32 < K)   // keep next weight chunk flowing into L2 (global_prefetch)
      __builtin_prefetch(&Wt[(size_t)(n0 + (tid & 127)) * K + k0 + 32], 0, 1);
    __syncthreads();
    v16bf a0 = lds_a_frag(&As[wm * 32 + 0][0], 34, lane);
    v16bf a1 = lds_a_frag(&As[wm * 32 + 16][0], 34, lane);
#pragma unroll
    for (int ni = 0; ni < 4; ++ni) {
      v16bf b = lds_b_frag(&Bs[wn * 64 + ni * 16][0], 34, lane);
      acc[0][ni] = wmma_bf16(a0, b, acc[0][ni]);
      acc[1][ni] = wmma_bf16(a1, b, acc[1][ni]);
    }
  }
  const int nlo = lane & 15, mhi = (lane >> 4) << 3;
#pragma unroll
  for (int mi = 0; mi < 2; ++mi)
#pragma unroll
    for (int ni = 0; ni < 4; ++ni)
#pragma unroll
      for (int r = 0; r < 8; ++r)
        C[(size_t)(m0 + wm * 32 + mi * 16 + mhi + r) * N +
          (n0 + wn * 64 + ni * 16 + nlo)] = acc[mi][ni][r];
}

// ---------------------------------------------------------------------------
// Fused RMS-norm + RoPE (+ gain*scale for Q) + bf16 pack. One token per block.
// Wave handles one head (lane l owns dims l, l+32, l+64, l+96).
// Also writes V transposed [b][kv][dim][seq] for contiguous PV B-fragments.
// ---------------------------------------------------------------------------
__device__ inline void rms_rope_head(const float* __restrict__ src,
                                     unsigned short* __restrict__ dst,
                                     int lane, float cs, float sn, float gain) {
  float x0 = src[lane], x1 = src[lane + 32], x2 = src[lane + 64], x3 = src[lane + 96];
  float ss = x0 * x0 + x1 * x1 + x2 * x2 + x3 * x3;
#pragma unroll
  for (int d = 1; d < 32; d <<= 1) ss += __shfl_xor(ss, d, 32);
  float inv = rsqrtf(ss * (1.0f / 128.0f) + EPS_RMS) * gain; // gain folds in (linear)
  x0 *= inv; x1 *= inv; x2 *= inv; x3 *= inv;
  float r0 =  x0 * cs + x1 * sn;
  float r1 = -x0 * sn + x1 * cs;
  dst[lane]      = f2bf(r0);
  dst[lane + 32] = f2bf(r1);
  dst[lane + 64] = f2bf(x2);
  dst[lane + 96] = f2bf(x3);
}

__global__ __launch_bounds__(256) void normrope_kernel(
    const float* __restrict__ Qp, const float* __restrict__ Kp,
    const float* __restrict__ Vp, const float* __restrict__ qgain,
    unsigned short* __restrict__ Qa, unsigned short* __restrict__ Ka,
    unsigned short* __restrict__ Vt) {
  const int token = blockIdx.x;
  const int bb = token / SS, t = token % SS;
  const int w = threadIdx.x >> 5, lane = threadIdx.x & 31;
  // inv_freq[l] = 10000^(-l/32) = exp(-ln(10000)*l/32)
  float freq = __expf(-9.210340371976184f * (float)lane * (1.0f / 32.0f));
  float ang = (float)t * freq;
  float sn, cs;
  __sincosf(ang, &sn, &cs);

#pragma unroll
  for (int it = 0; it < 2; ++it) {   // 8 waves x 2 = 16 Q heads
    int h = w + it * 8;
    rms_rope_head(Qp + (size_t)token * DD + h * HD,
                  Qa + (((size_t)(bb * NH + h)) * SS + t) * HD,
                  lane, cs, sn, qgain[h] * ATT_SCALE);
  }
  if (w < 4) {                        // waves 0..3: K heads
    int h = w;
    rms_rope_head(Kp + (size_t)token * (NKV * HD) + h * HD,
                  Ka + (((size_t)(bb * NKV + h)) * SS + t) * HD,
                  lane, cs, sn, 1.0f);
  } else {                            // waves 4..7: V heads, transpose store
    int h = w - 4;
    const float* src = Vp + (size_t)token * (NKV * HD) + h * HD;
    unsigned short* dst = Vt + ((size_t)(bb * NKV + h)) * HD * SS;
#pragma unroll
    for (int j = 0; j < 4; ++j) {
      int d = lane + 32 * j;
      dst[(size_t)d * SS + t] = f2bf(src[d]);
    }
  }
}

// ---------------------------------------------------------------------------
// Flash attention, causal, GQA (4 q-heads per kv-head).
// Grid (SS/128, NH, BB); 256 threads = 8 waves; wave owns 16 query rows.
// K-tiles of 64 keys, TDM double-buffered: wave 0 issues tensor_load_to_lds
// for tile kt+1 while everyone computes tile kt; s_wait_tensorcnt retires
// only the current tile's two descriptors (TENSORcnt is in-order per wave).
// ---------------------------------------------------------------------------
__global__ __launch_bounds__(256) void attn_kernel(
    const unsigned short* __restrict__ Qa, const unsigned short* __restrict__ Ka,
    const unsigned short* __restrict__ Vt, unsigned short* __restrict__ Y) {
  __shared__ unsigned short Kbuf[2][64 * 132];  // [key][dim], padded stride 132
  __shared__ unsigned short Vbuf[2][128 * 68];  // [dim][key], padded stride 68
  __shared__ unsigned short Pbuf[128 * 66];     // P rows, wave-private 16-row slabs

  const int q0 = blockIdx.x * 128;
  const int h = blockIdx.y, bb = blockIdx.z;
  const int kvh = h >> 2;
  const int tid = threadIdx.x, w = tid >> 5, lane = tid & 31;
  const int nlo = lane & 15, hi = lane >> 4;

  const unsigned short* Qb = Qa + ((size_t)(bb * NH + h)) * SS * HD;
  const unsigned short* Kb = Ka + ((size_t)(bb * NKV + kvh)) * SS * HD;
  const unsigned short* Vb = Vt + ((size_t)(bb * NKV + kvh)) * HD * SS;

  // Q fragments for this wave's 16 rows (4 chunks covering head_dim 128)
  BF16x16 qf[4];
  {
    const unsigned short* p = Qb + (size_t)(q0 + w * 16 + nlo) * HD;
    int koff = hi << 3;
#pragma unroll
    for (int kk = 0; kk < 4; ++kk)
#pragma unroll
      for (int v = 0; v < 8; ++v)
        qf[kk].w[v] =
            *(const unsigned int*)(p + kk * 32 + ((v & 4) << 2) + koff + ((v & 3) << 1));
  }

  float Mr[8], Lr[8];
  v8f o[8];
#pragma unroll
  for (int r = 0; r < 8; ++r) { Mr[r] = -1e30f; Lr[r] = 0.f; }
#pragma unroll
  for (int of = 0; of < 8; ++of)
#pragma unroll
    for (int r = 0; r < 8; ++r) o[of][r] = 0.f;

  const int ntiles = q0 / 64 + 2;   // keys 0 .. q0+127

#if HAVE_TDM
  if (w == 0) {   // prologue: DMA tile 0 into buffer 0
    tdm_load_2d((unsigned)(size_t)&Kbuf[0][0], Kb, HD, SS, HD, HD, 64, 5, 1);
    tdm_load_2d((unsigned)(size_t)&Vbuf[0][0], Vb, SS, HD, SS, 64, HD, 4, 1);
  }
#endif

  for (int kt = 0; kt < ntiles; ++kt) {
    const int cur = kt & 1;
    const unsigned short* Ks = Kbuf[cur];
    const unsigned short* Vs = Vbuf[cur];
    __syncthreads();   // everyone done reading buffer cur^1 (iteration kt-1)
#if HAVE_TDM
    if (w == 0) {
      if (kt + 1 < ntiles) {
        int nb = cur ^ 1;
        tdm_load_2d((unsigned)(size_t)&Kbuf[nb][0],
                    Kb + (size_t)(kt + 1) * 64 * HD, HD, SS, HD, HD, 64, 5, 1);
        tdm_load_2d((unsigned)(size_t)&Vbuf[nb][0],
                    Vb + (kt + 1) * 64, SS, HD, SS, 64, HD, 4, 1);
        __builtin_amdgcn_s_wait_tensorcnt(2);  // tile kt complete; kt+1 in flight
      } else {
        __builtin_amdgcn_s_wait_tensorcnt(0);
      }
    }
#else
    {
      unsigned short* Kw = Kbuf[cur];
      unsigned short* Vw = Vbuf[cur];
#pragma unroll
      for (int i = 0; i < 16; ++i) {   // K tile: 64 keys x 128 dims
        int e = tid + i * 256;
        int r = e >> 6, c = (e & 63) << 1;
        *(unsigned int*)(Kw + r * 132 + c) =
            *(const unsigned int*)(Kb + (size_t)(kt * 64 + r) * HD + c);
      }
#pragma unroll
      for (int i = 0; i < 16; ++i) {   // V^T tile: 128 dims x 64 keys
        int e = tid + i * 256;
        int r = e >> 5, c = (e & 31) << 1;
        *(unsigned int*)(Vw + r * 68 + c) =
            *(const unsigned int*)(Vb + (size_t)r * SS + kt * 64 + c);
      }
    }
#endif
    __syncthreads();   // tile kt visible to all waves

    // S = (scaled Q) @ K^T  : 4 n-blocks x 4 k-chunks of WMMA
    v8f sf[4];
#pragma unroll
    for (int ni = 0; ni < 4; ++ni) {
#pragma unroll
      for (int r = 0; r < 8; ++r) sf[ni][r] = 0.f;
#pragma unroll
      for (int kk = 0; kk < 4; ++kk) {
        v16bf b = lds_b_frag(Ks + ni * 16 * 132 + kk * 32, 132, lane);
        sf[ni] = wmma_bf16(qf[kk].v, b, sf[ni]);
      }
      if (kt * 64 + 63 > q0 + w * 16) {   // tile straddles the diagonal
        int kg = kt * 64 + ni * 16 + nlo;
        int qg = q0 + w * 16 + (hi << 3);
#pragma unroll
        for (int r = 0; r < 8; ++r)
          if (kg > qg + r) sf[ni][r] = -1e30f;
      }
    }

    // Online softmax: row m = r + 8*hi lives on 16 lanes -> shfl_xor(1,2,4,8)
#pragma unroll
    for (int r = 0; r < 8; ++r) {
      float mx = fmaxf(fmaxf(sf[0][r], sf[1][r]), fmaxf(sf[2][r], sf[3][r]));
#pragma unroll
      for (int d2 = 1; d2 < 16; d2 <<= 1) mx = fmaxf(mx, __shfl_xor(mx, d2, 32));
      float mn = fmaxf(Mr[r], mx);
      float ps = 0.f;
#pragma unroll
      for (int ni = 0; ni < 4; ++ni) {
        float p = __expf(sf[ni][r] - mn);
        sf[ni][r] = p;
        ps += p;
      }
#pragma unroll
      for (int d2 = 1; d2 < 16; d2 <<= 1) ps += __shfl_xor(ps, d2, 32);
      float fac = __expf(Mr[r] - mn);
      Lr[r] = Lr[r] * fac + ps;
      Mr[r] = mn;
#pragma unroll
      for (int of = 0; of < 8; ++of) o[of][r] *= fac;
    }

    // P (C-layout) -> LDS -> A-layout. Rows w*16..w*16+15 are wave-private,
    // so only a wave-local LDS RAW fence is needed.
#pragma unroll
    for (int ni = 0; ni < 4; ++ni)
#pragma unroll
      for (int r = 0; r < 8; ++r)
        Pbuf[(w * 16 + (hi << 3) + r) * 66 + ni * 16 + nlo] = f2bf(sf[ni][r]);
    asm volatile("s_wait_dscnt 0" ::: "memory");

    // O += P @ V^T(tile) : 8 dim-blocks x 2 key-chunks
#pragma unroll
    for (int of = 0; of < 8; ++of) {
#pragma unroll
      for (int kk = 0; kk < 2; ++kk) {
        BF16x16 af;
        const unsigned short* p = Pbuf + (w * 16 + nlo) * 66 + kk * 32;
        int koff = hi << 3;
#pragma unroll
        for (int v = 0; v < 8; ++v)
          af.w[v] = *(const unsigned int*)(p + ((v & 4) << 2) + koff + ((v & 3) << 1));
        v16bf bv = lds_b_frag(Vs + of * 16 * 68 + kk * 32, 68, lane);
        o[of] = wmma_bf16(af.v, bv, o[of]);
      }
    }
  }

  // Epilogue: divide by l, write bf16 into y [b][s][h][hd]
#pragma unroll
  for (int r = 0; r < 8; ++r) Lr[r] = 1.0f / Lr[r];
  unsigned short* Yb = Y + ((size_t)bb * SS) * DD + (size_t)h * HD;
#pragma unroll
  for (int of = 0; of < 8; ++of)
#pragma unroll
    for (int r = 0; r < 8; ++r) {
      int m = q0 + w * 16 + (hi << 3) + r;
      Yb[(size_t)m * DD + of * 16 + nlo] = f2bf(o[of][r] * Lr[r]);
    }
}

// ---------------------------------------------------------------------------
// Launch: 3 projection GEMMs -> norm/rope -> flash attention -> out GEMM
// ---------------------------------------------------------------------------
extern "C" void kernel_launch(void* const* d_in, const int* in_sizes, int n_in,
                              void* d_out, int out_size, void* d_ws, size_t ws_size,
                              hipStream_t stream) {
  (void)in_sizes; (void)n_in; (void)out_size; (void)ws_size;
  const float* x     = (const float*)d_in[0];
  const float* q_w   = (const float*)d_in[1];
  const float* k_w   = (const float*)d_in[2];
  const float* v_w   = (const float*)d_in[3];
  const float* out_w = (const float*)d_in[4];
  const float* qgain = (const float*)d_in[5];
  float* out = (float*)d_out;

  char* ws = (char*)d_ws;
  size_t off = 0;
  float* Qp = (float*)(ws + off); off += (size_t)BB * SS * DD * 4;        // 32 MB
  float* Kp = (float*)(ws + off); off += (size_t)BB * SS * NKV * HD * 4;  //  8 MB
  float* Vp = (float*)(ws + off); off += (size_t)BB * SS * NKV * HD * 4;  //  8 MB
  unsigned short* Qa = (unsigned short*)(ws + off); off += (size_t)BB * NH  * SS * HD * 2;
  unsigned short* Ka = (unsigned short*)(ws + off); off += (size_t)BB * NKV * SS * HD * 2;
  unsigned short* Vt = (unsigned short*)(ws + off); off += (size_t)BB * NKV * SS * HD * 2;
  unsigned short* Yh = (unsigned short*)(ws + off); off += (size_t)BB * SS * DD * 2;

  const int Mrows = BB * SS;  // 4096
  dim3 blk(256, 1, 1);

  gemm_kernel<false><<<dim3(DD / 128, Mrows / 128, 1), blk, 0, stream>>>(
      x, q_w, Qp, Mrows, DD, DD);
  gemm_kernel<false><<<dim3((NKV * HD) / 128, Mrows / 128, 1), blk, 0, stream>>>(
      x, k_w, Kp, Mrows, NKV * HD, DD);
  gemm_kernel<false><<<dim3((NKV * HD) / 128, Mrows / 128, 1), blk, 0, stream>>>(
      x, v_w, Vp, Mrows, NKV * HD, DD);

  normrope_kernel<<<dim3(Mrows, 1, 1), blk, 0, stream>>>(Qp, Kp, Vp, qgain, Qa, Ka, Vt);

  attn_kernel<<<dim3(SS / 128, NH, BB), blk, 0, stream>>>(Qa, Ka, Vt, Yh);

  gemm_kernel<true><<<dim3(DD / 128, Mrows / 128, 1), blk, 0, stream>>>(
      Yh, out_w, out, Mrows, DD, DD);
}
